// GraphEncoder_48275432407741
// MI455X (gfx1250) — matrix operand order repaired and checked by reading, not compiled, not run on hardware
//
#include <hip/hip_runtime.h>

// ---------------------------------------------------------------------------
// MI455X (gfx1250) heterogeneous-GCN encoder.
//   - Dense per-relation GEMMs in bf16 via v_wmma_f32_16x16x32_bf16 (fp32 acc)
//   - One wave computes a 16xN strip: A fragments loaded ONCE per k-step,
//     8 independent accumulators (no WMMA->WMMA RAW hazard, full unroll)
//   - Norms folded into edge scatter (c = n_src[s]*n_dst[d]); one wave32/edge
//   - Weights pre-packed into WMMA B-fragment lane layout (ISA 7.12.2)
//   - Edge scoring: wave32/edge dot-64 with __shfl_xor reduction
// ---------------------------------------------------------------------------

#define GDIN 128
#define WB_STRIDE 8192  // uints per packed 128x128 bf16 weight (128*128/2)

typedef __attribute__((ext_vector_type(16))) __bf16      v16bf;
typedef __attribute__((ext_vector_type(8)))  float       v8f;
typedef __attribute__((ext_vector_type(4)))  unsigned int u32x4;

__device__ __forceinline__ unsigned short f2bf(float f) {
  __bf16 b = (__bf16)f;          // RNE convert on gfx1250
  unsigned short u;
  __builtin_memcpy(&u, &b, 2);
  return u;
}

// -------------------------------- utility ----------------------------------
__global__ void fill_f32_k(float* p, float v, size_t n) {
  size_t stride = (size_t)gridDim.x * blockDim.x;
  for (size_t i = (size_t)blockIdx.x * blockDim.x + threadIdx.x; i < n; i += stride)
    p[i] = v;
}

__global__ void cvt_bf16_k(const float* __restrict__ x, unsigned short* __restrict__ y, size_t n) {
  size_t stride = (size_t)gridDim.x * blockDim.x;
  for (size_t i = (size_t)blockIdx.x * blockDim.x + threadIdx.x; i < n; i += stride)
    y[i] = f2bf(x[i]);
}

__global__ void degree_k(const int* __restrict__ src, const int* __restrict__ dst, int E,
                         float* du, float* di) {
  int stride = gridDim.x * blockDim.x;
  for (int e = blockIdx.x * blockDim.x + threadIdx.x; e < E; e += stride) {
    atomicAdd(&du[src[e]], 1.0f);
    atomicAdd(&di[dst[e]], 1.0f);
  }
}

__global__ void rsqrt_k(float* d, int n) {
  int i = blockIdx.x * blockDim.x + threadIdx.x;
  if (i < n) d[i] = rsqrtf(fmaxf(d[i], 1.0f));
}

// ----------------------- weight pack: fp32 -> B-fragment --------------------
// Packed order: idx = ((kt*NT + nt)*32 + lane)*8 + v
//   n = nt*16 + (lane&15);  k = kt*32 + 8*(lane>>4) + 16*(v>>2) + 2*(v&3)   (k,k+1 in one u32)
__global__ void pack_w_k(const float* __restrict__ W, unsigned int* __restrict__ out,
                         int K, int N) {
  int NT = N >> 4;
  int total = (K >> 5) * NT * 256;
  int stride = gridDim.x * blockDim.x;
  for (int idx = blockIdx.x * blockDim.x + threadIdx.x; idx < total; idx += stride) {
    int v    = idx & 7;
    int lane = (idx >> 3) & 31;
    int tile = idx >> 8;
    int nt   = tile % NT;
    int kt   = tile / NT;
    int n = nt * 16 + (lane & 15);
    int k = kt * 32 + 8 * (lane >> 4) + 16 * (v >> 2) + 2 * (v & 3);
    unsigned int lo = f2bf(W[(size_t)k * N + n]);
    unsigned int hi = f2bf(W[(size_t)(k + 1) * N + n]);
    out[idx] = lo | (hi << 16);
  }
}

// ------------------------------ bf16 WMMA GEMM ------------------------------
// C[M, NT*16] = A[M, KK](bf16) @ Bpacked(bf16 fragments), fp32 accumulate.
// One wave per 16-row strip covering ALL NT column tiles: A traffic is paid
// once (L2-bound roofline), B is tiny and L0-resident. KK/NT compile-time ->
// the whole loop nest unrolls into KK/32 * NT back-to-back WMMAs with
// independent accumulators.
template <int KK, int NT>
__global__ void gemm_bf16_wmma(const unsigned short* __restrict__ A,
                               const unsigned int* __restrict__ Bp,
                               float* __restrict__ C, int M) {
  const int lane  = threadIdx.x & 31;
  const int wave  = threadIdx.x >> 5;
  const int mtile = blockIdx.x * 8 + wave;
  if (mtile >= (M >> 4)) return;           // wave-uniform: EXEC stays all-ones

  const int kh = lane >> 4;                // lane half selects K sub-range
  const unsigned short* arow = A + (size_t)(mtile * 16 + (lane & 15)) * KK + kh * 8;
  const unsigned int*   bl   = Bp + lane * 8;

  v8f zero = {};
  v8f acc[NT];
#pragma unroll
  for (int nt = 0; nt < NT; ++nt) acc[nt] = zero;

#pragma unroll
  for (int kt = 0; kt < KK; kt += 32) {
    // A fragment (16-bit A 16x32 layout): two contiguous 16B runs per lane
    u32x4 a0 = *(const u32x4*)(arow + kt);
    u32x4 a1 = *(const u32x4*)(arow + kt + 16);
    v16bf av;
    __builtin_memcpy(&av, &a0, 16);
    __builtin_memcpy((char*)&av + 16, &a1, 16);
    const unsigned int* bk = bl + (size_t)(kt >> 5) * NT * 256;
#pragma unroll
    for (int nt = 0; nt < NT; ++nt) {
      u32x4 b0 = *(const u32x4*)(bk + nt * 256);
      u32x4 b1 = *(const u32x4*)(bk + nt * 256 + 4);
      v16bf bv;
      __builtin_memcpy(&bv, &b0, 16);
      __builtin_memcpy((char*)&bv + 16, &b1, 16);
      acc[nt] = __builtin_amdgcn_wmma_f32_16x16x32_bf16(false, av, false, bv,
                                                        (short)0, acc[nt], false, false);
    }
  }
  // C layout: VGPR r -> row r + 8*(lane>>4), col = lane&15
  const int N = NT * 16;
  float* crow = C + (size_t)(mtile * 16 + kh * 8) * N + (lane & 15);
#pragma unroll
  for (int nt = 0; nt < NT; ++nt)
#pragma unroll
    for (int r = 0; r < 8; ++r) crow[(size_t)r * N + nt * 16] = acc[nt][r];
}

// ------------------------- edge scatter (both directions) -------------------
// One wave per edge. c = nsrc[s]*ndst[d] is the symmetric GCN norm; folds the
// pre/post normalization of BOTH the forward and reverse relation.
template <int D>
__global__ void scatter_k(const int* __restrict__ src, const int* __restrict__ dst, int E,
                          const float* __restrict__ msrc, const float* __restrict__ mdst,
                          const float* __restrict__ nsrc, const float* __restrict__ ndst,
                          float* __restrict__ agg_dst, float* __restrict__ agg_src) {
  const int lane = threadIdx.x & 31;
  const int e = blockIdx.x * 8 + (threadIdx.x >> 5);
  if (e >= E) return;
  const int s = src[e], d = dst[e];
  const float c = nsrc[s] * ndst[d];
  constexpr int PER = D / 32;
  const float* ms = msrc + (size_t)s * D + lane * PER;
  const float* md = mdst + (size_t)d * D + lane * PER;
  float* ad = agg_dst + (size_t)d * D + lane * PER;
  float* as = agg_src + (size_t)s * D + lane * PER;
#pragma unroll
  for (int j = 0; j < PER; ++j) {
    atomicAdd(&ad[j], ms[j] * c);   // forward relation: item accumulates from user
    atomicAdd(&as[j], md[j] * c);   // reverse relation: user accumulates from item
  }
}

// ------------------------- finalize: bias (+relu) + bf16 --------------------
__global__ void finalize_k(const float* __restrict__ agg, const float* __restrict__ b0,
                           const float* __restrict__ b1, int D,
                           float* xf, unsigned short* xb, size_t n, int relu) {
  size_t stride = (size_t)gridDim.x * blockDim.x;
  for (size_t i = (size_t)blockIdx.x * blockDim.x + threadIdx.x; i < n; i += stride) {
    int f = (int)(i & (size_t)(D - 1));
    float v = agg[i] + b0[f];
    if (b1) v += b1[f];
    if (relu) v = fmaxf(v, 0.0f);
    if (xf) xf[i] = v;
    xb[i] = f2bf(v);
  }
}

// --------------------------- edge scoring (dot-64) --------------------------
__global__ void score_k(const int* __restrict__ src, const int* __restrict__ dst, int E,
                        const float* __restrict__ P, const float* __restrict__ I,
                        float* __restrict__ out) {
  const int lane = threadIdx.x & 31;
  const int e = blockIdx.x * 8 + (threadIdx.x >> 5);
  if (e >= E) return;
  const int s = src[e], d = dst[e];
  float2 a = ((const float2*)(P + (size_t)s * 64))[lane];
  float2 b = ((const float2*)(I + (size_t)d * 64))[lane];
  float v = a.x * b.x + a.y * b.y;
#pragma unroll
  for (int off = 16; off > 0; off >>= 1) v += __shfl_xor(v, off, 32);
  if (lane == 0) out[e] = v;
}

// ============================== orchestration ===============================
extern "C" void kernel_launch(void* const* d_in, const int* in_sizes, int n_in,
                              void* d_out, int out_size, void* d_ws, size_t ws_size,
                              hipStream_t stream) {
  (void)n_in; (void)out_size; (void)ws_size;
  const float* emb_u  = (const float*)d_in[0];
  const float* emb_i1 = (const float*)d_in[1];
  const float* emb_i2 = (const float*)d_in[2];
  const float* W_in   = (const float*)d_in[3];
  const float* b_in   = (const float*)d_in[4];
  const float* W_hid  = (const float*)d_in[5];
  const float* b_hid  = (const float*)d_in[6];
  const float* W_out  = (const float*)d_in[7];
  const float* b_out  = (const float*)d_in[8];
  const float* Wp1    = (const float*)d_in[9];
  const float* Wp2    = (const float*)d_in[10];
  const int* src1 = (const int*)d_in[11];
  const int* dst1 = (const int*)d_in[12];
  const int* src2 = (const int*)d_in[13];
  const int* dst2 = (const int*)d_in[14];
  float* out = (float*)d_out;

  const int NU = in_sizes[0] / GDIN;
  const int N1 = in_sizes[1] / GDIN;
  const int N2 = in_sizes[2] / GDIN;
  const int E  = in_sizes[11];

  // ---- workspace bump allocator (256B aligned) ----
  char* base = (char*)d_ws;
  size_t off = 0;
  auto alloc = [&](size_t bytes) -> char* {
    char* p = base + off;
    off += (bytes + 255) & ~(size_t)255;
    return p;
  };
  float* nu1 = (float*)alloc((size_t)NU * 4);
  float* ni1 = (float*)alloc((size_t)N1 * 4);
  float* nu2 = (float*)alloc((size_t)NU * 4);
  float* ni2 = (float*)alloc((size_t)N2 * 4);
  float* m_u0 = (float*)alloc((size_t)NU * GDIN * 4);
  float* m_u2 = (float*)alloc((size_t)NU * GDIN * 4);
  float* m_i1 = (float*)alloc((size_t)N1 * GDIN * 4);
  float* m_i2 = (float*)alloc((size_t)N2 * GDIN * 4);
  float* agg_u  = (float*)alloc((size_t)NU * GDIN * 4);
  float* agg_i1 = (float*)alloc((size_t)N1 * GDIN * 4);
  float* agg_i2 = (float*)alloc((size_t)N2 * GDIN * 4);
  float* xi1_f = (float*)alloc((size_t)N1 * 64 * 4);   // final item embeddings (fp32)
  float* xi2_f = (float*)alloc((size_t)N2 * 64 * 4);
  unsigned short* xb_u  = (unsigned short*)alloc((size_t)NU * GDIN * 2);
  unsigned short* xb_i1 = (unsigned short*)alloc((size_t)N1 * GDIN * 2);
  unsigned short* xb_i2 = (unsigned short*)alloc((size_t)N2 * GDIN * 2);
  unsigned int* wb   = (unsigned int*)alloc((size_t)4 * WB_STRIDE * 4);
  unsigned int* wpb1 = (unsigned int*)alloc((size_t)2048 * 4);
  unsigned int* wpb2 = (unsigned int*)alloc((size_t)2048 * 4);
  float* p1 = m_u0;  // reuse: m_* dead after the layer loop
  float* p2 = m_u2;

  auto fill = [&](float* p, size_t n) {
    fill_f32_k<<<2048, 256, 0, stream>>>(p, 0.0f, n);
  };
  auto pack = [&](const float* W, unsigned int* o, int K, int N) {
    int total = (K >> 5) * (N >> 4) * 256;
    pack_w_k<<<(total + 255) / 256, 256, 0, stream>>>(W, o, K, N);
  };
  auto gemm128 = [&](const unsigned short* A, const unsigned int* Bp, float* C,
                     int M, int Dout) {
    dim3 grid((M / 16 + 7) / 8);
    if (Dout == 128)
      gemm_bf16_wmma<128, 8><<<grid, 256, 0, stream>>>(A, Bp, C, M);
    else
      gemm_bf16_wmma<128, 4><<<grid, 256, 0, stream>>>(A, Bp, C, M);
  };
  const int eblocks = (E + 7) / 8;

  // ---- symmetric GCN degree norms ----
  fill(nu1, NU); fill(ni1, N1); fill(nu2, NU); fill(ni2, N2);
  degree_k<<<4096, 256, 0, stream>>>(src1, dst1, E, nu1, ni1);
  degree_k<<<4096, 256, 0, stream>>>(src2, dst2, E, nu2, ni2);
  rsqrt_k<<<(NU + 255) / 256, 256, 0, stream>>>(nu1, NU);
  rsqrt_k<<<(N1 + 255) / 256, 256, 0, stream>>>(ni1, N1);
  rsqrt_k<<<(NU + 255) / 256, 256, 0, stream>>>(nu2, NU);
  rsqrt_k<<<(N2 + 255) / 256, 256, 0, stream>>>(ni2, N2);

  // ---- inputs -> bf16 A operands ----
  cvt_bf16_k<<<2048, 256, 0, stream>>>(emb_u,  xb_u,  (size_t)NU * GDIN);
  cvt_bf16_k<<<2048, 256, 0, stream>>>(emb_i1, xb_i1, (size_t)N1 * GDIN);
  cvt_bf16_k<<<2048, 256, 0, stream>>>(emb_i2, xb_i2, (size_t)N2 * GDIN);

  // ---- three GCN layers ----
  const float* Ws[3] = {W_in, W_hid, W_out};
  const float* Bs[3] = {b_in, b_hid, b_out};
  const int Douts[3] = {128, 128, 64};
  for (int l = 0; l < 3; ++l) {
    const int Dout = Douts[l];
    for (int r = 0; r < 4; ++r)
      pack(Ws[l] + (size_t)r * GDIN * Dout, wb + r * WB_STRIDE, GDIN, Dout);

    gemm128(xb_u,  wb + 0 * WB_STRIDE, m_u0, NU, Dout);  // user->d1 relation
    gemm128(xb_i1, wb + 1 * WB_STRIDE, m_i1, N1, Dout);  // d1->user relation
    gemm128(xb_u,  wb + 2 * WB_STRIDE, m_u2, NU, Dout);  // user->d2 relation
    gemm128(xb_i2, wb + 3 * WB_STRIDE, m_i2, N2, Dout);  // d2->user relation

    fill(agg_u, (size_t)NU * Dout);
    fill(agg_i1, (size_t)N1 * Dout);
    fill(agg_i2, (size_t)N2 * Dout);

    if (Dout == 128) {
      scatter_k<128><<<eblocks, 256, 0, stream>>>(src1, dst1, E, m_u0, m_i1, nu1, ni1, agg_i1, agg_u);
      scatter_k<128><<<eblocks, 256, 0, stream>>>(src2, dst2, E, m_u2, m_i2, nu2, ni2, agg_i2, agg_u);
    } else {
      scatter_k<64><<<eblocks, 256, 0, stream>>>(src1, dst1, E, m_u0, m_i1, nu1, ni1, agg_i1, agg_u);
      scatter_k<64><<<eblocks, 256, 0, stream>>>(src2, dst2, E, m_u2, m_i2, nu2, ni2, agg_i2, agg_u);
    }

    const int relu = (l < 2);
    finalize_k<<<2048, 256, 0, stream>>>(agg_u, Bs[l] + 1 * Dout, Bs[l] + 3 * Dout, Dout,
                                         nullptr, xb_u, (size_t)NU * Dout, relu);
    finalize_k<<<2048, 256, 0, stream>>>(agg_i1, Bs[l] + 0 * Dout, nullptr, Dout,
                                         (l == 2) ? xi1_f : nullptr, xb_i1, (size_t)N1 * Dout, relu);
    finalize_k<<<2048, 256, 0, stream>>>(agg_i2, Bs[l] + 2 * Dout, nullptr, Dout,
                                         (l == 2) ? xi2_f : nullptr, xb_i2, (size_t)N2 * Dout, relu);
  }

  // ---- per-domain projection of user embeddings (K=64, N=64) ----
  pack(Wp1, wpb1, 64, 64);
  pack(Wp2, wpb2, 64, 64);
  {
    dim3 grid((NU / 16 + 7) / 8);
    gemm_bf16_wmma<64, 4><<<grid, 256, 0, stream>>>(xb_u, wpb1, p1, NU);
    gemm_bf16_wmma<64, 4><<<grid, 256, 0, stream>>>(xb_u, wpb2, p2, NU);
  }

  // ---- edge scores: [s1 ; s2] ----
  score_k<<<eblocks, 256, 0, stream>>>(src1, dst1, E, p1, xi1_f, out);
  score_k<<<eblocks, 256, 0, stream>>>(src2, dst2, E, p2, xi2_f, out + E);
}
